// ESM3s_28879360098450
// MI455X (gfx1250) — matrix operand order repaired and checked by reading, not compiled, not run on hardware
//
#include <hip/hip_runtime.h>

#define S_LEN  2048
#define DMODEL 1024
#define FDIM   2560
#define NHEAD  16
#define HDIM   64

typedef __bf16 v16bf __attribute__((ext_vector_type(16)));
typedef __bf16 v8bf  __attribute__((ext_vector_type(8)));
typedef float  v8f   __attribute__((ext_vector_type(8)));

// ---------------------------------------------------------------------------
// WMMA helpers (ISA 7.12.2 layouts, wave32)
// ---------------------------------------------------------------------------
static __device__ inline v8f wmma_bf16(v16bf a, v16bf b, v8f c) {
    return __builtin_amdgcn_wmma_f32_16x16x32_bf16(
        /*neg_a=*/false, a, /*neg_b=*/false, b,
        /*c_mod=*/(short)0, c, /*reuse_a=*/false, /*reuse_b=*/false);
}

static __device__ inline v8f zero8() {
    v8f z;
#pragma unroll
    for (int i = 0; i < 8; ++i) z[i] = 0.0f;
    return z;
}

// xor-lane swizzle via ds_swizzle (group-of-32: and=0x1f, or=0, xor=MASK).
// MASK must be an ICE -> template parameter.
template <int MASK>
static __device__ inline float swz_xor(float v) {
    int i = __builtin_amdgcn_ds_swizzle(__float_as_int(v), (MASK << 10) | 31);
    return __int_as_float(i);
}

static __device__ inline float red_add32(float v) {
    v += swz_xor<16>(v); v += swz_xor<8>(v); v += swz_xor<4>(v);
    v += swz_xor<2>(v);  v += swz_xor<1>(v);
    return v;
}
static __device__ inline float red_max16(float v) {
    v = fmaxf(v, swz_xor<8>(v)); v = fmaxf(v, swz_xor<4>(v));
    v = fmaxf(v, swz_xor<2>(v)); v = fmaxf(v, swz_xor<1>(v));
    return v;
}
static __device__ inline float red_add16(float v) {
    v += swz_xor<8>(v); v += swz_xor<4>(v);
    v += swz_xor<2>(v); v += swz_xor<1>(v);
    return v;
}

// A fragment: 16(M) x 32(K), row-major source with leading dim ld.
// lanes 0-15: V0..3 = K0..7, V4..7 = K16..23 ; lanes 16-31: K8..15, K24..31
static __device__ inline v16bf load_frag_a(const __bf16* base, int ld, int row0, int k0) {
    const int lane = threadIdx.x & 31;
    const int r  = lane & 15;
    const int hi = lane >> 4;
    const __bf16* p = base + (size_t)(row0 + r) * ld + k0 + hi * 8;
    v8bf lo = *(const v8bf*)(p);
    v8bf hh = *(const v8bf*)(p + 16);
    v16bf out;
#pragma unroll
    for (int i = 0; i < 8; ++i) { out[i] = lo[i]; out[i + 8] = hh[i]; }
    return out;
}

// B fragment: 32(K) x 16(N) from row-major [N,K] source (computing A @ B^T).
// lane n (0-15): K0..7, K16..23 ; lane n+16: K8..15, K24..31
static __device__ inline v16bf load_frag_b(const __bf16* base, int ld, int col0, int k0) {
    const int lane = threadIdx.x & 31;
    const int n  = lane & 15;
    const int hi = lane >> 4;
    const __bf16* p = base + (size_t)(col0 + n) * ld + k0 + hi * 8;
    v8bf lo = *(const v8bf*)(p);
    v8bf hh = *(const v8bf*)(p + 16);
    v16bf out;
#pragma unroll
    for (int i = 0; i < 8; ++i) { out[i] = lo[i]; out[i + 8] = hh[i]; }
    return out;
}

// async 16B global -> LDS copy (ASYNCcnt-tracked); inline asm is portable
// across toolchains (builtin arity differs between clang-22 / clang-23).
static __device__ inline void async_copy_b128(unsigned int lds_off, const void* gptr) {
    unsigned long long ga = (unsigned long long)(uintptr_t)gptr;
    asm volatile("global_load_async_to_lds_b128 %0, %1, off"
                 :: "v"(lds_off), "v"(ga) : "memory");
}
static __device__ inline void wait_asynccnt0() {
    asm volatile("s_wait_asynccnt 0x0" ::: "memory");
}

// ---------------------------------------------------------------------------
// Weight conversion / packing (f32 -> bf16)
// ---------------------------------------------------------------------------
__global__ void conv_bf16_kernel(const float* __restrict__ a, __bf16* __restrict__ o, long n) {
    long i = (long)blockIdx.x * blockDim.x + threadIdx.x;
    if (i < n) o[i] = (__bf16)a[i];
}

__global__ void pack_qkv_kernel(const float* __restrict__ wq, const float* __restrict__ wk,
                                const float* __restrict__ wv, __bf16* __restrict__ o, long nn) {
    long i = (long)blockIdx.x * blockDim.x + threadIdx.x;  // nn = D*D
    if (i >= 3 * nn) return;
    float v;
    if (i < nn)            v = wq[i];
    else if (i < 2 * nn)   v = wk[i - nn];
    else                   v = wv[i - 2 * nn];
    o[i] = (__bf16)v;
}

// interleave g/u rows in 16-row groups: packed row 32b+t (t<16 -> g row 16b+t, else u row F+16b+t-16)
__global__ void pack_w12_kernel(const float* __restrict__ w12, __bf16* __restrict__ o) {
    long i = (long)blockIdx.x * blockDim.x + threadIdx.x;  // over 2F*D
    if (i >= (long)2 * FDIM * DMODEL) return;
    int rp = (int)(i / DMODEL);
    int c  = (int)(i % DMODEL);
    int b  = rp >> 5;
    int t  = rp & 31;
    int src = (t < 16) ? (b * 16 + t) : (FDIM + b * 16 + (t - 16));
    o[i] = (__bf16)w12[(size_t)src * DMODEL + c];
}

// ---------------------------------------------------------------------------
// LayerNorm (f32 in -> bf16 out), one block per row
// ---------------------------------------------------------------------------
__global__ __launch_bounds__(256) void ln_kernel(const float* __restrict__ x,
                                                 const float* __restrict__ w,
                                                 const float* __restrict__ b,
                                                 __bf16* __restrict__ out, int D) {
    const int row  = blockIdx.x;
    const int lane = threadIdx.x & 31;
    const int wv   = threadIdx.x >> 5;
    const float* xr = x + (size_t)row * D;
    float s = 0.f, sq = 0.f;
    for (int i = threadIdx.x; i < D; i += 256) { float v = xr[i]; s += v; sq += v * v; }
    s  = red_add32(s);
    sq = red_add32(sq);
    __shared__ float ss[8], sg[8];
    if (lane == 0) { ss[wv] = s; sg[wv] = sq; }
    __syncthreads();
    float tot = 0.f, tq = 0.f;
#pragma unroll
    for (int i = 0; i < 8; ++i) { tot += ss[i]; tq += sg[i]; }
    float mu  = tot / D;
    float var = tq / D - mu * mu;
    float rs  = rsqrtf(var + 1e-5f);
    for (int i = threadIdx.x; i < D; i += 256) {
        float y = (xr[i] - mu) * rs * w[i] + b[i];
        out[(size_t)row * D + i] = (__bf16)y;
    }
}

// ---------------------------------------------------------------------------
// Q/K layernorm + RoPE + head reshape; V transpose. One block per seq pos.
// qkv: f32 [S, 3D] -> qh/kh bf16 [H][S][64], vT bf16 [H][64][S]
// ---------------------------------------------------------------------------
__global__ __launch_bounds__(256) void qkrope_kernel(const float* __restrict__ qkv,
                                                     const float* __restrict__ qw,
                                                     const float* __restrict__ kw,
                                                     const float* __restrict__ cosc,
                                                     const float* __restrict__ sinc,
                                                     __bf16* __restrict__ qh,
                                                     __bf16* __restrict__ kh,
                                                     __bf16* __restrict__ vT) {
    const int s    = blockIdx.x;
    const int lane = threadIdx.x & 31;
    const int wv   = threadIdx.x >> 5;
    const float* base = qkv + (size_t)s * (3 * DMODEL);
    __shared__ float buf[DMODEL];
    __shared__ float red[32];

    float sq = 0.f, sqq = 0.f, sk = 0.f, skk = 0.f;
    for (int i = threadIdx.x; i < DMODEL; i += 256) {
        float a = base[i];          sq += a;  sqq += a * a;
        float b = base[DMODEL + i]; sk += b;  skk += b * b;
    }
    sq  = red_add32(sq);
    sqq = red_add32(sqq);
    sk  = red_add32(sk);
    skk = red_add32(skk);
    if (lane == 0) { red[wv] = sq; red[8 + wv] = sqq; red[16 + wv] = sk; red[24 + wv] = skk; }
    __syncthreads();
    float tq = 0.f, tqq = 0.f, tk = 0.f, tkk = 0.f;
#pragma unroll
    for (int i = 0; i < 8; ++i) { tq += red[i]; tqq += red[8 + i]; tk += red[16 + i]; tkk += red[24 + i]; }
    const float muq = tq / DMODEL, rsq_ = rsqrtf(tqq / DMODEL - muq * muq + 1e-5f);
    const float muk = tk / DMODEL, rsk_ = rsqrtf(tkk / DMODEL - muk * muk + 1e-5f);

    // ---- Q: LN -> LDS -> rope -> store
    for (int i = threadIdx.x; i < DMODEL; i += 256)
        buf[i] = (base[i] - muq) * rsq_ * qw[i];
    __syncthreads();
    for (int i = threadIdx.x; i < DMODEL; i += 256) {
        int dh = i & 63, h = i >> 6;
        float c = cosc[(size_t)s * 64 + dh], sn = sinc[(size_t)s * 64 + dh];
        float rot = (dh < 32) ? -buf[i + 32] : buf[i - 32];
        qh[(size_t)h * S_LEN * 64 + (size_t)s * 64 + dh] = (__bf16)(buf[i] * c + rot * sn);
    }
    __syncthreads();
    // ---- K
    for (int i = threadIdx.x; i < DMODEL; i += 256)
        buf[i] = (base[DMODEL + i] - muk) * rsk_ * kw[i];
    __syncthreads();
    for (int i = threadIdx.x; i < DMODEL; i += 256) {
        int dh = i & 63, h = i >> 6;
        float c = cosc[(size_t)s * 64 + dh], sn = sinc[(size_t)s * 64 + dh];
        float rot = (dh < 32) ? -buf[i + 32] : buf[i - 32];
        kh[(size_t)h * S_LEN * 64 + (size_t)s * 64 + dh] = (__bf16)(buf[i] * c + rot * sn);
    }
    // ---- V transposed: vT[h][d][s]
    for (int i = threadIdx.x; i < DMODEL; i += 256) {
        int dh = i & 63, h = i >> 6;
        vT[(size_t)h * 64 * S_LEN + (size_t)dh * S_LEN + s] = (__bf16)base[2 * DMODEL + i];
    }
}

// ---------------------------------------------------------------------------
// Generic bf16 WMMA GEMM: out = epilogue(A[M,K] @ B[N,K]^T)
// block = 256 thr (8 waves); wave tile 32(M) x 64(N); grid = (N/64, M/256)
// B tile (64x32 bf16 = 4KB) staged in LDS via async global->LDS copies,
// double-buffered: next tile's ASYNC copy overlaps current tile's WMMAs.
// EPI 0: f32 store   EPI 1: f32 resid add   EPI 2: silu(g)*u -> bf16 (packed N)
// ---------------------------------------------------------------------------
template <int EPI>
__global__ __launch_bounds__(256) void gemm_kernel(const __bf16* __restrict__ A,
                                                   const __bf16* __restrict__ B,
                                                   int M, int N, int K,
                                                   float* __restrict__ outF,
                                                   const float* __restrict__ resid,
                                                   __bf16* __restrict__ outBf, int Fd) {
    __shared__ __bf16 bsm[2][64 * 32];   // 2 x 4KB B tiles
    const int tid  = threadIdx.x;
    const int wv   = tid >> 5;
    const int lane = tid & 31;
    const int m0 = blockIdx.y * 256 + wv * 32;
    const int n0 = blockIdx.x * 64;

    // cooperative B-tile staging: 256 threads x 16B = 4KB
    const int srow   = tid >> 2;   // 0..63
    const int schunk = tid & 3;    // 0..3 (8 bf16 each)
    const __bf16* Bg = B + (size_t)(n0 + srow) * K + schunk * 8;
    const unsigned int lds0 =
        (unsigned int)(uintptr_t)(&bsm[0][srow * 32 + schunk * 8]);
    const unsigned int lds1 =
        (unsigned int)(uintptr_t)(&bsm[1][srow * 32 + schunk * 8]);

    v8f acc[2][4];
#pragma unroll
    for (int im = 0; im < 2; ++im)
#pragma unroll
        for (int j = 0; j < 4; ++j) acc[im][j] = zero8();

    const int T = K / 32;
    async_copy_b128(lds0, Bg);                         // prologue: tile 0
    for (int i = 0; i < T; ++i) {
        const int b  = i & 1;
        const int k0 = i * 32;
        wait_asynccnt0();      // my writes for tile i have landed in LDS
        __syncthreads();       // tile i visible; tile i-1 buffer fully consumed
        if (i + 1 < T)         // overlap tile i+1 copy with tile i compute
            async_copy_b128(b ? lds0 : lds1, Bg + (k0 + 32));

        v16bf a0 = load_frag_a(A, K, m0, k0);
        v16bf a1 = load_frag_a(A, K, m0 + 16, k0);
#pragma unroll
        for (int j = 0; j < 4; ++j) {
            v16bf bb = load_frag_b(bsm[b], 32, j * 16, 0);
            acc[0][j] = wmma_bf16(a0, bb, acc[0][j]);
            acc[1][j] = wmma_bf16(a1, bb, acc[1][j]);
        }
    }

    const int nl = lane & 15, hi = lane >> 4;
    if (EPI == 2) {
#pragma unroll
        for (int im = 0; im < 2; ++im)
#pragma unroll
            for (int j = 0; j < 4; j += 2) {
                int fcol = (((n0 + j * 16) >> 5) << 4) + nl;
#pragma unroll
                for (int r = 0; r < 8; ++r) {
                    int row = m0 + im * 16 + 8 * hi + r;
                    float g = acc[im][j][r];
                    float u = acc[im][j + 1][r];
                    float sg = g / (1.0f + __expf(-g));
                    outBf[(size_t)row * Fd + fcol] = (__bf16)(sg * u);
                }
            }
    } else {
#pragma unroll
        for (int im = 0; im < 2; ++im)
#pragma unroll
            for (int j = 0; j < 4; ++j) {
                int col = n0 + j * 16 + nl;
#pragma unroll
                for (int r = 0; r < 8; ++r) {
                    int row = m0 + im * 16 + 8 * hi + r;
                    float v = acc[im][j][r];
                    if (EPI == 1) v += resid[(size_t)row * N + col];
                    outF[(size_t)row * N + col] = v;
                }
            }
    }
}

// ---------------------------------------------------------------------------
// Flash attention (non-causal). grid=(S/128, H), block=256 (8 waves x 16 q).
// qh/kh: bf16 [H][S][64]; vT: bf16 [H][64][S]; ctx: bf16 [S][D]
// ---------------------------------------------------------------------------
__global__ __launch_bounds__(256) void attn_kernel(const __bf16* __restrict__ qh,
                                                   const __bf16* __restrict__ kh,
                                                   const __bf16* __restrict__ vT,
                                                   __bf16* __restrict__ ctx) {
    __shared__ __bf16 smem[8 * 16 * 32];
    const int wv   = threadIdx.x >> 5;
    const int lane = threadIdx.x & 31;
    const int nl = lane & 15, hi = lane >> 4;
    const int h  = blockIdx.y;
    const int q0 = blockIdx.x * 128 + wv * 16;

    const __bf16* Q  = qh + (size_t)h * S_LEN * 64;
    const __bf16* Kp = kh + (size_t)h * S_LEN * 64;
    const __bf16* Vp = vT + (size_t)h * 64 * S_LEN;
    __bf16* myP = smem + wv * (16 * 32);

    v16bf qa0 = load_frag_a(Q, 64, q0, 0);
    v16bf qa1 = load_frag_a(Q, 64, q0, 32);

    float mrow[8], lrow[8];
    v8f o[4];
#pragma unroll
    for (int r = 0; r < 8; ++r) { mrow[r] = -3.0e38f; lrow[r] = 0.f; }
#pragma unroll
    for (int j = 0; j < 4; ++j) o[j] = zero8();

    const float sc = (1.0f / 64.0f) * 1.44269504088896f;  // scale * log2(e)

    for (int kb = 0; kb < S_LEN; kb += 32) {
        v8f s0 = zero8(), s1 = zero8();
        s0 = wmma_bf16(qa0, load_frag_b(Kp, 64, kb, 0), s0);
        s0 = wmma_bf16(qa1, load_frag_b(Kp, 64, kb, 32), s0);
        s1 = wmma_bf16(qa0, load_frag_b(Kp, 64, kb + 16, 0), s1);
        s1 = wmma_bf16(qa1, load_frag_b(Kp, 64, kb + 16, 32), s1);

#pragma unroll
        for (int r = 0; r < 8; ++r) {
            float t0 = s0[r] * sc, t1 = s1[r] * sc;
            float mx = red_max16(fmaxf(t0, t1));
            float newm = fmaxf(mrow[r], mx);
            float corr = exp2f(mrow[r] - newm);
            float p0 = exp2f(t0 - newm);
            float p1 = exp2f(t1 - newm);
            float sum = red_add16(p0 + p1);
            lrow[r] = lrow[r] * corr + sum;
            mrow[r] = newm;
#pragma unroll
            for (int j = 0; j < 4; ++j) o[j][r] *= corr;
            // C-layout (row = r + 8*hi, col = nl) -> LDS row-major [16][32] bf16
            myP[(r + 8 * hi) * 32 + nl]      = (__bf16)p0;
            myP[(r + 8 * hi) * 32 + nl + 16] = (__bf16)p1;
        }
        asm volatile("s_wait_dscnt 0x0" ::: "memory");
        v16bf pa = load_frag_a((const __bf16*)myP, 32, 0, 0);
#pragma unroll
        for (int j = 0; j < 4; ++j)
            o[j] = wmma_bf16(pa, load_frag_b(Vp, S_LEN, j * 16, kb), o[j]);
    }

#pragma unroll
    for (int j = 0; j < 4; ++j) {
        int col = h * 64 + j * 16 + nl;
#pragma unroll
        for (int r = 0; r < 8; ++r) {
            int row = q0 + r + 8 * hi;
            float v = o[j][r] / lrow[r];
            ctx[(size_t)row * DMODEL + col] = (__bf16)v;
        }
    }
}

// ---------------------------------------------------------------------------
// Launch
// ---------------------------------------------------------------------------
extern "C" void kernel_launch(void* const* d_in, const int* in_sizes, int n_in,
                              void* d_out, int out_size, void* d_ws, size_t ws_size,
                              hipStream_t stream) {
    const float* x     = (const float*)d_in[0];
    const float* cosc  = (const float*)d_in[1];
    const float* sinc  = (const float*)d_in[2];
    const float* ln1_w = (const float*)d_in[3];
    const float* ln1_b = (const float*)d_in[4];
    const float* wq    = (const float*)d_in[5];
    const float* wk    = (const float*)d_in[6];
    const float* wv    = (const float*)d_in[7];
    const float* qln_w = (const float*)d_in[8];
    const float* kln_w = (const float*)d_in[9];
    const float* wo    = (const float*)d_in[10];
    const float* ln2_w = (const float*)d_in[11];
    const float* ln2_b = (const float*)d_in[12];
    const float* w12   = (const float*)d_in[13];
    const float* w2    = (const float*)d_in[14];
    float* out = (float*)d_out;

    const int S = S_LEN, D = DMODEL, F = FDIM;
    char* ws = (char*)d_ws;
    size_t off = 0;
    auto alloc = [&](size_t bytes) { char* p = ws + off; off = (off + bytes + 255) & ~(size_t)255; return p; };

    __bf16* hx    = (__bf16*)alloc((size_t)S * D * 2);       // also reused for ln2 output
    __bf16* wqkv  = (__bf16*)alloc((size_t)3 * D * D * 2);
    __bf16* wo_b  = (__bf16*)alloc((size_t)D * D * 2);
    __bf16* w12p  = (__bf16*)alloc((size_t)2 * F * D * 2);
    __bf16* w2_b  = (__bf16*)alloc((size_t)D * F * 2);
    float*  qkv   = (float*) alloc((size_t)S * 3 * D * 4);   // reused for act (bf16, smaller)
    __bf16* qh    = (__bf16*)alloc((size_t)S * D * 2);
    __bf16* kh    = (__bf16*)alloc((size_t)S * D * 2);
    __bf16* vT    = (__bf16*)alloc((size_t)S * D * 2);
    __bf16* ctx   = (__bf16*)alloc((size_t)S * D * 2);
    float*  x1    = (float*) alloc((size_t)S * D * 4);
    __bf16* act   = (__bf16*)qkv;                            // reuse: qkv dead after rope stage

    // 1. weight conversion / packing (f32 -> bf16)
    {
        long n = (long)3 * D * D;
        pack_qkv_kernel<<<(int)((n + 255) / 256), 256, 0, stream>>>(wq, wk, wv, wqkv, (long)D * D);
        n = (long)D * D;
        conv_bf16_kernel<<<(int)((n + 255) / 256), 256, 0, stream>>>(wo, wo_b, n);
        n = (long)2 * F * D;
        pack_w12_kernel<<<(int)((n + 255) / 256), 256, 0, stream>>>(w12, w12p);
        n = (long)D * F;
        conv_bf16_kernel<<<(int)((n + 255) / 256), 256, 0, stream>>>(w2, w2_b, n);
    }

    // 2. ln1: x -> hx (bf16)
    ln_kernel<<<S, 256, 0, stream>>>(x, ln1_w, ln1_b, hx, D);

    // 3. qkv = hx @ wqkv^T  (f32 out for accurate q/k layernorm)
    gemm_kernel<0><<<dim3(3 * D / 64, S / 256), 256, 0, stream>>>(
        hx, wqkv, S, 3 * D, D, qkv, nullptr, nullptr, 0);

    // 4. q/k LN + rope -> qh, kh ; v transpose -> vT
    qkrope_kernel<<<S, 256, 0, stream>>>(qkv, qln_w, kln_w, cosc, sinc, qh, kh, vT);

    // 5. flash attention -> ctx (bf16 [S,D])
    attn_kernel<<<dim3(S / 128, NHEAD), 256, 0, stream>>>(qh, kh, vT, ctx);

    // 6. x1 = x + ctx @ wo^T
    gemm_kernel<1><<<dim3(D / 64, S / 256), 256, 0, stream>>>(
        ctx, wo_b, S, D, D, x1, x, nullptr, 0);

    // 7. ln2: x1 -> hx (bf16)
    ln_kernel<<<S, 256, 0, stream>>>(x1, ln2_w, ln2_b, hx, D);

    // 8. act = silu(g)*u with packed w12 -> bf16 [S,F]
    gemm_kernel<2><<<dim3(2 * F / 64, S / 256), 256, 0, stream>>>(
        hx, w12p, S, 2 * F, D, nullptr, nullptr, act, F);

    // 9. out = x1 + act @ w2^T
    gemm_kernel<1><<<dim3(D / 64, S / 256), 256, 0, stream>>>(
        act, w2_b, S, D, F, out, x1, nullptr, 0);
}
